// DeepKoopman_49366354100597
// MI455X (gfx1250) — matrix operand order repaired and checked by reading, not compiled, not run on hardware
//
#include <hip/hip_runtime.h>
#include <math.h>

// ---------------------------------------------------------------------------
// Types for CDNA5 WMMA (gfx1250, wave32)
// ---------------------------------------------------------------------------
typedef __attribute__((ext_vector_type(16))) _Float16 v16h;
typedef __attribute__((ext_vector_type(8)))  _Float16 v8h;
typedef __attribute__((ext_vector_type(8)))  float    v8f;

#define DT_C 0.02f
#define OUT_TSTRIDE 4224   // 33*128 floats per batch row

// ---------------------------------------------------------------------------
// Fragment loaders (per CDNA5 ISA 7.12.2 layouts)
// A 16x32 f16 (MxK): lane L holds row m=L&15; halves: K in [hi*8, hi*8+8) and
// [16+hi*8, 16+hi*8+8) where hi=L>>4.  Two 16B LDS loads.
// B 32x16 f16 (KxN): lane L holds column n=L&15; K in [hi*8, hi*8+8) and
// [16+hi*8, 16+hi*8+8) where hi=L>>4 (same K order as A).
// ---------------------------------------------------------------------------
__device__ __forceinline__ v16h load_a(const _Float16* tile, int ld, int lane) {
    int m  = lane & 15;
    int hi = lane >> 4;
    const _Float16* p = tile + m * ld + hi * 8;
    union { v16h v; v8h h[2]; } u;
    u.h[0] = *(const v8h*)(p);
    u.h[1] = *(const v8h*)(p + 16);
    return u.v;
}

__device__ __forceinline__ v16h load_b(const _Float16* tile, int ld, int lane) {
    const int n = lane & 15, hi = lane >> 4;
    const _Float16* p = tile + (hi * 8) * ld + n;
    v16h v;
#pragma unroll
    for (int s = 0; s < 8; ++s) {
        v[s]     = p[s * ld];
        v[s + 8] = p[(16 + s) * ld];
    }
    return v;
}

__device__ __forceinline__ v8f wmma_f16(v16h a, v16h b, v8f c) {
    return __builtin_amdgcn_wmma_f32_16x16x32_f16(false, a, false, b,
                                                  (short)0, c, false, false);
}

// ---------------------------------------------------------------------------
// Shared GEMM: C(128x256) = relu(A(128x256) @ W(256x256) + bias), in-place on
// sAct.  Wave w owns rows [16w,16w+16): reads/writes only its own rows, so no
// barrier is needed (A fragments are cached in VGPRs before the first write).
// sAct row stride 264 f16, W row stride 264 f16.
// ---------------------------------------------------------------------------
__device__ __forceinline__ void gemm256_relu_inplace(
    _Float16* sAct, const _Float16* sW, const float* sBias, int lane, int wid)
{
    const int Mb = wid << 4;
    const int hi = lane >> 4, n = lane & 15;
    v16h aF[8];
#pragma unroll
    for (int k = 0; k < 8; ++k)
        aF[k] = load_a(sAct + Mb * 264 + k * 32, 264, lane);
    for (int nt = 0; nt < 16; ++nt) {
        float bv = sBias[nt * 16 + n];
        v8f acc = {0.f, 0.f, 0.f, 0.f, 0.f, 0.f, 0.f, 0.f};
#pragma unroll
        for (int k = 0; k < 8; ++k)
            acc = wmma_f16(aF[k], load_b(sW + (k * 32) * 264 + nt * 16, 264, lane), acc);
#pragma unroll
        for (int j = 0; j < 8; ++j) {
            float v = acc[j] + bv;
            sAct[(Mb + j + hi * 8) * 264 + nt * 16 + n] = (_Float16)(v > 0.f ? v : 0.f);
        }
    }
}

// ---------------------------------------------------------------------------
// Kernel 1: encoder.  y0 = MLP3(x[:,0,:]) stashed into out[:,32,0:5]
// (rollout reads the stash before decode t=32 overwrites it).
// ---------------------------------------------------------------------------
__global__ __launch_bounds__(256) void encoder_kernel(
    const float* __restrict__ x,
    const float* __restrict__ We1, const float* __restrict__ be1,
    const float* __restrict__ We2, const float* __restrict__ be2,
    const float* __restrict__ We3, const float* __restrict__ be3,
    float* __restrict__ outp)
{
    __shared__ __attribute__((aligned(16))) _Float16 sWe1[128 * 264];
    __shared__ __attribute__((aligned(16))) _Float16 sWe2[256 * 264];
    __shared__ __attribute__((aligned(16))) _Float16 sX[128 * 136];
    __shared__ __attribute__((aligned(16))) _Float16 sH[128 * 264];
    __shared__ float sWe3f[256 * 5];
    __shared__ float sBe1[256];
    __shared__ float sBe2[256];
    __shared__ float sBe3[5];

    const int tid  = threadIdx.x;
    const int lane = tid & 31;
    const int wid  = tid >> 5;
    const int rowBase = blockIdx.x * 128;

    for (int i = tid; i < 128 * 256; i += 256)
        sWe1[(i >> 8) * 264 + (i & 255)] = (_Float16)We1[i];
    for (int i = tid; i < 256 * 256; i += 256)
        sWe2[(i >> 8) * 264 + (i & 255)] = (_Float16)We2[i];
    for (int i = tid; i < 256 * 5; i += 256) sWe3f[i] = We3[i];
    if (tid < 256) { sBe1[tid] = be1[tid]; sBe2[tid] = be2[tid]; }
    if (tid < 5) sBe3[tid] = be3[tid];
    // x[:,0,:] tile -> f16 LDS (row stride 136)
    for (int i = tid; i < 128 * 128; i += 256) {
        int r = i >> 7, c = i & 127;
        sX[r * 136 + c] = (_Float16)x[(size_t)(rowBase + r) * (50 * 128) + c];
    }
    __syncthreads();

    const int Mb = wid << 4;
    const int hi = lane >> 4, n = lane & 15;
    // L1: (128x128) @ We1(128x256), K=128 -> 4 fragments
    {
        v16h aF[4];
#pragma unroll
        for (int k = 0; k < 4; ++k)
            aF[k] = load_a(sX + Mb * 136 + k * 32, 136, lane);
        for (int nt = 0; nt < 16; ++nt) {
            float bv = sBe1[nt * 16 + n];
            v8f acc = {0.f, 0.f, 0.f, 0.f, 0.f, 0.f, 0.f, 0.f};
#pragma unroll
            for (int k = 0; k < 4; ++k)
                acc = wmma_f16(aF[k], load_b(sWe1 + (k * 32) * 264 + nt * 16, 264, lane), acc);
#pragma unroll
            for (int j = 0; j < 8; ++j) {
                float v = acc[j] + bv;
                sH[(Mb + j + hi * 8) * 264 + nt * 16 + n] = (_Float16)(v > 0.f ? v : 0.f);
            }
        }
    }
    // L2: in-place, wave-private rows, no barrier needed
    gemm256_relu_inplace(sH, sWe2, sBe2, lane, wid);
    __syncthreads();   // L3 reads arbitrary rows
    // L3: 256 -> 5 (VALU reduction), stash y0 in out[:,32,0:5]
    if (tid < 128) {
        float acc[5];
#pragma unroll
        for (int j = 0; j < 5; ++j) acc[j] = sBe3[j];
        const _Float16* hrow = sH + tid * 264;
        for (int k = 0; k < 256; ++k) {
            float h = (float)hrow[k];
#pragma unroll
            for (int j = 0; j < 5; ++j) acc[j] += h * sWe3f[k * 5 + j];
        }
        float* dst = outp + (size_t)(rowBase + tid) * OUT_TSTRIDE + 32 * 128;
#pragma unroll
        for (int j = 0; j < 5; ++j) dst[j] = acc[j];
    }
}

// ---------------------------------------------------------------------------
// Decode current state sY -> out[:, t, :].  L1 (K=5) in VALU, L2/L3 in WMMA.
// ---------------------------------------------------------------------------
__device__ __forceinline__ void decode_store(
    const float* sY, const float* sWd1f, const float* sBd1,
    const _Float16* sWd2, const float* sBd2,
    const _Float16* sWd3, const float* sBd3,
    _Float16* sH, float* outp, int rowBase, int t)
{
    const int tid  = threadIdx.x;
    const int lane = tid & 31;
    const int wid  = tid >> 5;

    // L1: H1[r][c] = relu(sum_{k<5} y[r][k]*Wd1[k][c] + bd1[c])
    {
        int r  = tid >> 1;
        int c0 = (tid & 1) << 7;
        float y0 = sY[r * 8 + 0], y1 = sY[r * 8 + 1], y2 = sY[r * 8 + 2];
        float y3 = sY[r * 8 + 3], y4 = sY[r * 8 + 4];
        for (int c = c0; c < c0 + 128; ++c) {
            float a = sBd1[c] + y0 * sWd1f[c] + y1 * sWd1f[256 + c]
                    + y2 * sWd1f[512 + c] + y3 * sWd1f[768 + c]
                    + y4 * sWd1f[1024 + c];
            sH[r * 264 + c] = (_Float16)(a > 0.f ? a : 0.f);
        }
    }
    __syncthreads();

    // L2: 256x256 WMMA, in-place (wave-private rows)
    gemm256_relu_inplace(sH, sWd2, sBd2, lane, wid);

    // L3: (128x256) @ Wd3(256x128); stage f32 into own rows, coalesced store
    {
        const int Mb = wid << 4;
        const int hi = lane >> 4, n = lane & 15;
        v16h aF[8];
#pragma unroll
        for (int k = 0; k < 8; ++k)
            aF[k] = load_a(sH + Mb * 264 + k * 32, 264, lane);
        v8f acc[8];
#pragma unroll
        for (int nt = 0; nt < 8; ++nt) {
            acc[nt] = (v8f){0.f, 0.f, 0.f, 0.f, 0.f, 0.f, 0.f, 0.f};
#pragma unroll
            for (int k = 0; k < 8; ++k)
                acc[nt] = wmma_f16(aF[k], load_b(sWd3 + (k * 32) * 136 + nt * 16, 136, lane), acc[nt]);
        }
#pragma unroll
        for (int nt = 0; nt < 8; ++nt) {
            float bv = sBd3[nt * 16 + n];
#pragma unroll
            for (int j = 0; j < 8; ++j) {
                int row = Mb + j + hi * 8;
                float* hrow = (float*)((char*)sH + row * 528);
                hrow[nt * 16 + n] = acc[nt][j] + bv;
            }
        }
        // 512B fully-coalesced store per row
        for (int r16 = 0; r16 < 16; ++r16) {
            int row = Mb + r16;
            const float4* src = (const float4*)((const char*)sH + row * 528);
            float4 v = src[lane];
            float* dst = outp + (size_t)(rowBase + row) * OUT_TSTRIDE + t * 128 + lane * 4;
            *(float4*)dst = v;
        }
    }
}

// ---------------------------------------------------------------------------
// Kernel 2: fused 33-decode rollout.  All weights LDS-resident (~308 KB).
// ---------------------------------------------------------------------------
__global__ __launch_bounds__(256) void rollout_kernel(
    const float* __restrict__ Wd1, const float* __restrict__ bd1,
    const float* __restrict__ Wd2, const float* __restrict__ bd2,
    const float* __restrict__ Wd3, const float* __restrict__ bd3,
    const float* __restrict__ Wc1, const float* __restrict__ bc1,
    const float* __restrict__ Wc2, const float* __restrict__ bc2,
    const float* __restrict__ Wc3, const float* __restrict__ bc3,
    const float* __restrict__ Wr1, const float* __restrict__ br1,
    const float* __restrict__ Wr2, const float* __restrict__ br2,
    const float* __restrict__ Wr3, const float* __restrict__ br3,
    float* __restrict__ outp)
{
    __shared__ __attribute__((aligned(16))) _Float16 sWd2[256 * 264];
    __shared__ __attribute__((aligned(16))) _Float16 sWd3[256 * 136];
    __shared__ __attribute__((aligned(16))) _Float16 sH[128 * 264];
    __shared__ __attribute__((aligned(16))) _Float16 sWc2[2 * 64 * 72];
    __shared__ __attribute__((aligned(16))) _Float16 sWr2[64 * 72];
    __shared__ float sWd1f[5 * 256];
    __shared__ float sBd1[256];
    __shared__ float sBd2[256];
    __shared__ float sBd3[128];
    __shared__ float sWc1[128];   // (P,1,64)
    __shared__ float sBc1[128];   // (P,64)
    __shared__ float sBc2[128];   // (P,64)
    __shared__ float sWc3[256];   // (P,64,2)
    __shared__ float sBc3[4];     // (P,2)
    __shared__ float sWr1[64];
    __shared__ float sBr1[64];
    __shared__ float sBr2[64];
    __shared__ float sWr3[64];
    __shared__ float sBr3[1];
    __shared__ float sY[128 * 8]; // y[0..4], rad0@5, rad1@6

    const int tid  = threadIdx.x;
    const int lane = tid & 31;
    const int wid  = tid >> 5;
    const int rowBase = blockIdx.x * 128;

    // omega staging overlays the decoder activation buffer (disjoint in time)
    float*    sh2 = (float*)sH;                          // 128 x 65 f32
    _Float16* sh1 = (_Float16*)((char*)sH + 33280);      // 128 x 72 f16

    // ---- load weights into LDS --------------------------------------------
    for (int i = tid; i < 256 * 256; i += 256)
        sWd2[(i >> 8) * 264 + (i & 255)] = (_Float16)Wd2[i];
    for (int i = tid; i < 256 * 128; i += 256)
        sWd3[(i >> 7) * 136 + (i & 127)] = (_Float16)Wd3[i];
    for (int i = tid; i < 2 * 64 * 64; i += 256) {
        int p = i >> 12, rem = i & 4095;
        sWc2[p * (64 * 72) + (rem >> 6) * 72 + (rem & 63)] = (_Float16)Wc2[i];
    }
    for (int i = tid; i < 64 * 64; i += 256)
        sWr2[(i >> 6) * 72 + (i & 63)] = (_Float16)Wr2[i];
    for (int i = tid; i < 5 * 256; i += 256) sWd1f[i] = Wd1[i];
    if (tid < 256) { sBd1[tid] = bd1[tid]; sBd2[tid] = bd2[tid]; sWc3[tid] = Wc3[tid]; }
    if (tid < 128) { sBd3[tid] = bd3[tid]; sWc1[tid] = Wc1[tid];
                     sBc1[tid] = bc1[tid]; sBc2[tid] = bc2[tid]; }
    if (tid < 64)  { sWr1[tid] = Wr1[tid]; sBr1[tid] = br1[tid];
                     sBr2[tid] = br2[tid]; sWr3[tid] = Wr3[tid]; }
    if (tid < 4)   sBc3[tid] = bc3[tid];
    if (tid == 0)  sBr3[0] = br3[0];
    // initial state from encoder stash (out[:,32,0:5])
    if (tid < 128) {
        const float* src = outp + (size_t)(rowBase + tid) * OUT_TSTRIDE + 32 * 128;
#pragma unroll
        for (int j = 0; j < 5; ++j) sY[tid * 8 + j] = src[j];
    }
    __syncthreads();

    // ---- t = 0 decode ------------------------------------------------------
    decode_store(sY, sWd1f, sBd1, sWd2, sBd2, sWd3, sBd3, sH, outp, rowBase, 0);

    // ---- 32 rollout steps --------------------------------------------------
    for (int t = 1; t <= 32; ++t) {
        // radii of complex pairs
        if (tid < 128) {
            float a0 = sY[tid * 8 + 0], a1 = sY[tid * 8 + 1];
            float a2 = sY[tid * 8 + 2], a3 = sY[tid * 8 + 3];
            sY[tid * 8 + 5] = sqrtf(a0 * a0 + a1 * a1);
            sY[tid * 8 + 6] = sqrtf(a2 * a2 + a3 * a3);
        }
        __syncthreads();

        float freq0 = 0.f, mu0 = 0.f, freq1 = 0.f, mu1 = 0.f, orr = 0.f;
        for (int net = 0; net < 3; ++net) {
            // L1 outer product: scalar -> 64, relu, write f16 tile
            {
                int r  = tid >> 1;
                int n0 = (tid & 1) << 5;
                float s = (net == 0) ? sY[r * 8 + 5]
                        : (net == 1) ? sY[r * 8 + 6]
                                     : sY[r * 8 + 4];
                const float* W1 = (net == 2) ? sWr1 : sWc1 + net * 64;
                const float* B1 = (net == 2) ? sBr1 : sBc1 + net * 64;
                for (int nn = n0; nn < n0 + 32; ++nn) {
                    float h = s * W1[nn] + B1[nn];
                    sh1[r * 72 + nn] = (_Float16)(h > 0.f ? h : 0.f);
                }
            }
            __syncthreads();
            // L2: (128x64) @ (64x64) WMMA, relu -> sh2 (f32, own rows)
            {
                const _Float16* W2 = (net == 2) ? sWr2 : sWc2 + net * (64 * 72);
                const float*    B2 = (net == 2) ? sBr2 : sBc2 + net * 64;
                const int Mb = wid << 4;
                const int hi = lane >> 4, n = lane & 15;
                v16h aF0 = load_a(sh1 + Mb * 72, 72, lane);
                v16h aF1 = load_a(sh1 + Mb * 72 + 32, 72, lane);
#pragma unroll
                for (int nt = 0; nt < 4; ++nt) {
                    float bv = B2[nt * 16 + n];
                    v8f acc = {0.f, 0.f, 0.f, 0.f, 0.f, 0.f, 0.f, 0.f};
                    acc = wmma_f16(aF0, load_b(W2 + nt * 16, 72, lane), acc);
                    acc = wmma_f16(aF1, load_b(W2 + 32 * 72 + nt * 16, 72, lane), acc);
#pragma unroll
                    for (int j = 0; j < 8; ++j) {
                        float v = acc[j] + bv;
                        sh2[(Mb + j + hi * 8) * 65 + nt * 16 + n] = (v > 0.f ? v : 0.f);
                    }
                }
            }
            __syncthreads();
            // L3: 64 -> {2 or 1} reduction into per-thread registers
            if (tid < 128) {
                const float* h2 = sh2 + tid * 65;
                if (net < 2) {
                    float f = sBc3[net * 2 + 0], m = sBc3[net * 2 + 1];
                    const float* W3 = sWc3 + net * 128;
                    for (int k = 0; k < 64; ++k) {
                        float h = h2[k];
                        f += h * W3[k * 2];
                        m += h * W3[k * 2 + 1];
                    }
                    if (net == 0) { freq0 = f; mu0 = m; }
                    else          { freq1 = f; mu1 = m; }
                } else {
                    float o = sBr3[0];
                    for (int k = 0; k < 64; ++k) o += h2[k] * sWr3[k];
                    orr = o;
                }
            }
            // no barrier needed: next-net L1 writes sh1 (disjoint from sh2);
            // the following L1 barrier orders sh2 reads before next L2 writes.
        }

        // advance
        if (tid < 128) {
            float* y = sY + tid * 8;
            float s0 = expf(mu0 * DT_C), c0 = cosf(freq0 * DT_C), n0 = sinf(freq0 * DT_C);
            float s1 = expf(mu1 * DT_C), c1 = cosf(freq1 * DT_C), n1 = sinf(freq1 * DT_C);
            float a = y[0], b = y[1];
            y[0] = s0 * (a * c0 - b * n0);
            y[1] = s0 * (a * n0 + b * c0);
            float c = y[2], d = y[3];
            y[2] = s1 * (c * c1 - d * n1);
            y[3] = s1 * (c * n1 + d * c1);
            y[4] = y[4] * expf(orr * DT_C);
        }
        __syncthreads();

        decode_store(sY, sWd1f, sBd1, sWd2, sBd2, sWd3, sBd3, sH, outp, rowBase, t);
    }
}

// ---------------------------------------------------------------------------
// Host launcher
// ---------------------------------------------------------------------------
extern "C" void kernel_launch(void* const* d_in, const int* in_sizes, int n_in,
                              void* d_out, int out_size, void* d_ws, size_t ws_size,
                              hipStream_t stream) {
    (void)in_sizes; (void)n_in; (void)out_size; (void)d_ws; (void)ws_size;
    const float* x   = (const float*)d_in[0];
    const float* We1 = (const float*)d_in[1];
    const float* be1 = (const float*)d_in[2];
    const float* We2 = (const float*)d_in[3];
    const float* be2 = (const float*)d_in[4];
    const float* We3 = (const float*)d_in[5];
    const float* be3 = (const float*)d_in[6];
    const float* Wd1 = (const float*)d_in[7];
    const float* bd1 = (const float*)d_in[8];
    const float* Wd2 = (const float*)d_in[9];
    const float* bd2 = (const float*)d_in[10];
    const float* Wd3 = (const float*)d_in[11];
    const float* bd3 = (const float*)d_in[12];
    const float* Wc1 = (const float*)d_in[13];
    const float* bc1 = (const float*)d_in[14];
    const float* Wc2 = (const float*)d_in[15];
    const float* bc2 = (const float*)d_in[16];
    const float* Wc3 = (const float*)d_in[17];
    const float* bc3 = (const float*)d_in[18];
    const float* Wr1 = (const float*)d_in[19];
    const float* br1 = (const float*)d_in[20];
    const float* Wr2 = (const float*)d_in[21];
    const float* br2 = (const float*)d_in[22];
    const float* Wr3 = (const float*)d_in[23];
    const float* br3 = (const float*)d_in[24];
    float* outp = (float*)d_out;

    encoder_kernel<<<128, 256, 0, stream>>>(x, We1, be1, We2, be2, We3, be3, outp);
    rollout_kernel<<<128, 256, 0, stream>>>(Wd1, bd1, Wd2, bd2, Wd3, bd3,
                                            Wc1, bc1, Wc2, bc2, Wc3, bc3,
                                            Wr1, br1, Wr2, br2, Wr3, br3, outp);
}
